// NaMixedOp_4544075399253
// MI455X (gfx1250) — matrix operand order repaired and checked by reading, not compiled, not run on hardware
//
#include <hip/hip_runtime.h>
#include <hip/hip_bf16.h>

typedef __attribute__((ext_vector_type(16))) __bf16 bf16x16;
typedef __attribute__((ext_vector_type(8)))  float  floatx8;

#define GEMM_WAVES 4
#define LDS_STRIDE 132   // 128 + 4 pad (keeps float4 alignment: 132*4 = 33*16 bytes)

// ---------------------------------------------------------------------------
// zero a float region with float4 stores (grid-stride)
__global__ void zero_kernel(float4* __restrict__ p, long n4) {
    long i = (long)blockIdx.x * blockDim.x + threadIdx.x;
    long stride = (long)gridDim.x * blockDim.x;
    float4 z; z.x = 0.f; z.y = 0.f; z.z = 0.f; z.w = 0.f;
    for (; i < n4; i += stride) p[i] = z;
}

// ---------------------------------------------------------------------------
// Combine the 5 candidate-op weight matrices (scaled by DARTS weights) into
// bf16 WMMA B-fragments, pre-swizzled so the GEMM kernel loads 32 contiguous
// bytes per lane.  B-fragment layout for 32x16 bf16 tile (wave32):
//   lane<16 : elems j=0..7 -> k = kt*32+j,   j=8..15 -> k = kt*32+8+j   (n = nt*16+lane)
//   lane>=16: elems j=0..7 -> k = kt*32+8+j, j=8..15 -> k = kt*32+16+j  (n = nt*16+lane-16)
__global__ void prep_kernel(const float* __restrict__ wts,
    const float* __restrict__ Wgcn, const float* __restrict__ bgcn,
    const float* __restrict__ Wss,  const float* __restrict__ Wsn,
    const float* __restrict__ bsage,
    const float* __restrict__ Wg1,  const float* __restrict__ bg1,
    const float* __restrict__ Wg2,  const float* __restrict__ bg2,
    const float* __restrict__ Wgc1, const float* __restrict__ Wgc2,
    const float* __restrict__ bgc,
    const float* __restrict__ Wmlp, const float* __restrict__ bmlp,
    __bf16* __restrict__ WBf, __bf16* __restrict__ W1f, __bf16* __restrict__ W2f,
    float* __restrict__ btot, float* __restrict__ b1f)
{
    int idx = blockIdx.x * blockDim.x + threadIdx.x;
    float w0 = wts[0], w1 = wts[1], w2 = wts[2], w3 = wts[3], w4 = wts[4];
    if (idx < 16384) {
        // Wbig: K=256 (4 blocks of 64) x Nout=64 -> 8 kt x 4 nt tiles
        int t = idx >> 9, lane = (idx >> 4) & 31, j = idx & 15;
        int kt = t >> 2, nt = t & 3;
        int k = kt * 32 + (lane >> 4) * 8 + (j & 7) + (j >> 3) * 16;
        int n = nt * 16 + (lane & 15);
        int blk = k >> 6, kk = k & 63;
        float v;
        if      (blk == 0) v = w1 * Wss[kk * 64 + n] + w3 * Wgc1[kk * 64 + n] + w4 * Wmlp[kk * 64 + n];
        else if (blk == 1) v = w0 * Wgcn[kk * 64 + n];
        else if (blk == 2) v = w1 * Wsn[kk * 64 + n];
        else               v = w3 * Wgc2[kk * 64 + n];
        WBf[(t * 32 + lane) * 16 + j] = (__bf16)v;
    } else if (idx < 24576) {
        // W_gin1: 64 x 128 -> 2 kt x 8 nt tiles (unscaled)
        int i = idx - 16384;
        int t = i >> 9, lane = (i >> 4) & 31, j = i & 15;
        int kt = t >> 3, nt = t & 7;
        int k = kt * 32 + (lane >> 4) * 8 + (j & 7) + (j >> 3) * 16;
        int n = nt * 16 + (lane & 15);
        W1f[(t * 32 + lane) * 16 + j] = (__bf16)Wg1[k * 128 + n];
    } else if (idx < 32768) {
        // w2 * W_gin2: 128 x 64 -> 4 kt x 4 nt tiles
        int i = idx - 24576;
        int t = i >> 9, lane = (i >> 4) & 31, j = i & 15;
        int kt = t >> 2, nt = t & 3;
        int k = kt * 32 + (lane >> 4) * 8 + (j & 7) + (j >> 3) * 16;
        int n = nt * 16 + (lane & 15);
        W2f[(t * 32 + lane) * 16 + j] = (__bf16)(w2 * Wg2[k * 64 + n]);
    } else if (idx < 32768 + 64) {
        int n = idx - 32768;
        btot[n] = w0 * bgcn[n] + w1 * bsage[n] + w2 * bg2[n] + w3 * bgc[n] + w4 * bmlp[n];
    } else if (idx < 32768 + 64 + 128) {
        int n = idx - 32768 - 64;
        b1f[n] = bg1[n];
    }
}

// ---------------------------------------------------------------------------
// weighted in-degree: wdeg[dst] += ew[e]
__global__ void wdeg_kernel(const long long* __restrict__ ei,
                            const float* __restrict__ ew,
                            float* __restrict__ wdeg, int E)
{
    int e = blockIdx.x * blockDim.x + threadIdx.x;
    if (e < E) {
        int d = (int)ei[(size_t)E + e];
        atomicAdd(wdeg + d, ew[e]);
    }
}

// ---------------------------------------------------------------------------
// fused edge gather/scatter: one gather of x[src], two weighted scatter-adds.
// 16 threads per edge, float4 per thread.  x (12.8MB) + accumulators (25.6MB)
// stay resident in the 192MB L2, so the random gathers/atomics hit L2.
__global__ void edge_kernel(const float* __restrict__ x,
                            const long long* __restrict__ ei,
                            const float* __restrict__ ew,
                            const float* __restrict__ wdeg,
                            float* __restrict__ agg,
                            float* __restrict__ gcn_pre, int E)
{
    int t = blockIdx.x * blockDim.x + threadIdx.x;
    int e = t >> 4, q = t & 15;
    if (e >= E) return;
    int s = (int)ei[e];
    int d = (int)ei[(size_t)E + e];
    float w = ew[e];
    float norm = w * rsqrtf((wdeg[s] + 1.0f) * (wdeg[d] + 1.0f));
    const float4 xv = ((const float4*)x)[(size_t)s * 16 + q];
    float* ad = agg     + (size_t)d * 64 + q * 4;
    float* gd = gcn_pre + (size_t)d * 64 + q * 4;
    atomicAdd(ad + 0, w * xv.x); atomicAdd(ad + 1, w * xv.y);
    atomicAdd(ad + 2, w * xv.z); atomicAdd(ad + 3, w * xv.w);
    atomicAdd(gd + 0, norm * xv.x); atomicAdd(gd + 1, norm * xv.y);
    atomicAdd(gd + 2, norm * xv.z); atomicAdd(gd + 3, norm * xv.w);
}

// ---------------------------------------------------------------------------
__device__ inline void load8(const float* p, float* v) {
    float4 a = *(const float4*)(p);
    float4 b = *(const float4*)(p + 4);
    v[0] = a.x; v[1] = a.y; v[2] = a.z; v[3] = a.w;
    v[4] = b.x; v[5] = b.y; v[6] = b.z; v[7] = b.w;
}

// A-fragment pack for 16x32 bf16 A (wave32 layout):
//   lane<16 : elems 0..7 -> K = base+0..7,  elems 8..15 -> K = base+16..23
//   lane>=16: elems 0..7 -> K = base+8..15, elems 8..15 -> K = base+24..31
// (the +hi*8 shift is applied by the caller via the load offsets)
__device__ inline bf16x16 pack16(const float* u, const float* w) {
    bf16x16 a;
#pragma unroll
    for (int i = 0; i < 8; ++i) { a[i] = (__bf16)u[i]; a[8 + i] = (__bf16)w[i]; }
    return a;
}

// ---------------------------------------------------------------------------
// Fused node kernel: one 16-row tile per wave.
//   out = [x | gcn_pre + x/deg | agg/wdeg | agg] @ Wbig(256x64)
//       + relu(((1+eps)x + agg) @ Wgin1 + b1) @ (w2*Wgin2)  + btot
// All GEMMs via v_wmma_f32_16x16x32_bf16 (f32 accumulate).  GIN hidden is
// staged through padded LDS to convert C/D layout -> A-fragment layout.
__global__ void __launch_bounds__(GEMM_WAVES * 32)
gemm_kernel(const float* __restrict__ x, const float* __restrict__ wdeg,
            const float* __restrict__ gcn_pre, const float* __restrict__ agg,
            const __bf16* __restrict__ WBf, const __bf16* __restrict__ W1f,
            const __bf16* __restrict__ W2f, const float* __restrict__ btot,
            const float* __restrict__ b1f, const float* __restrict__ epsp,
            float* __restrict__ out, int N)
{
    __shared__ __align__(16) float lds[GEMM_WAVES * 16 * LDS_STRIDE];
    const int wid  = threadIdx.x >> 5;
    const int lane = threadIdx.x & 31;
    const int lm   = lane & 15;
    const int hi   = lane >> 4;
    const int m0   = (blockIdx.x * GEMM_WAVES + wid) * 16;
    const bool active = (m0 < N);
    float* slab = lds + wid * (16 * LDS_STRIDE);

    floatx8 acc[4] = {};
    const bf16x16* WBv = (const bf16x16*)WBf;
    const bf16x16* W1v = (const bf16x16*)W1f;
    const bf16x16* W2v = (const bf16x16*)W2f;

    if (active) {
        const float* xrow = x       + (size_t)(m0 + lm) * 64;
        const float* grow = gcn_pre + (size_t)(m0 + lm) * 64;
        const float* arow = agg     + (size_t)(m0 + lm) * 64;
        const float wd     = wdeg[m0 + lm];
        const float invdeg = 1.0f / (wd + 1.0f);
        const float invw   = 1.0f / (wd + 1e-16f);
        const float epl    = 1.0f + epsp[0];

        // ---- main combined GEMM, K = 256 ----
#pragma unroll
        for (int kt = 0; kt < 8; ++kt) {
            const int blk = kt >> 1;
            const int ko  = (kt & 1) * 32 + hi * 8;
            float u[8], v[8], t[8];
            if (blk == 0) {                       // x
                load8(xrow + ko, u); load8(xrow + ko + 16, v);
            } else if (blk == 1) {                // gcn_pre + x/deg
                load8(grow + ko, u); load8(grow + ko + 16, v);
                load8(xrow + ko, t);
#pragma unroll
                for (int i = 0; i < 8; ++i) u[i] += t[i] * invdeg;
                load8(xrow + ko + 16, t);
#pragma unroll
                for (int i = 0; i < 8; ++i) v[i] += t[i] * invdeg;
            } else if (blk == 2) {                // agg / wdeg  (SAGE mean)
                load8(arow + ko, u); load8(arow + ko + 16, v);
#pragma unroll
                for (int i = 0; i < 8; ++i) { u[i] *= invw; v[i] *= invw; }
            } else {                              // agg (GraphConv)
                load8(arow + ko, u); load8(arow + ko + 16, v);
            }
            const bf16x16 a = pack16(u, v);
#pragma unroll
            for (int nt = 0; nt < 4; ++nt)
                acc[nt] = __builtin_amdgcn_wmma_f32_16x16x32_bf16(
                    false, a, false, WBv[(kt * 4 + nt) * 32 + lane],
                    (short)0, acc[nt], false, false);
        }

        // ---- GIN layer 1: g = (1+eps)x + agg, K = 64, H = 128 ----
        floatx8 hid[8] = {};
#pragma unroll
        for (int kt = 0; kt < 2; ++kt) {
            const int ko = kt * 32 + hi * 8;
            float u[8], v[8], t[8];
            load8(xrow + ko, u); load8(arow + ko, t);
#pragma unroll
            for (int i = 0; i < 8; ++i) u[i] = epl * u[i] + t[i];
            load8(xrow + ko + 16, v); load8(arow + ko + 16, t);
#pragma unroll
            for (int i = 0; i < 8; ++i) v[i] = epl * v[i] + t[i];
            const bf16x16 a = pack16(u, v);
#pragma unroll
            for (int nt = 0; nt < 8; ++nt)
                hid[nt] = __builtin_amdgcn_wmma_f32_16x16x32_bf16(
                    false, a, false, W1v[(kt * 8 + nt) * 32 + lane],
                    (short)0, hid[nt], false, false);
        }
        // bias + relu, C/D layout -> row-major LDS
        // C/D: lane<16 holds (M=v, N=lane); lane>=16 holds (M=v+8, N=lane-16)
#pragma unroll
        for (int nt = 0; nt < 8; ++nt) {
            const int c = nt * 16 + lm;
            const float bb = b1f[c];
#pragma unroll
            for (int v2 = 0; v2 < 8; ++v2) {
                const float f = hid[nt][v2] + bb;
                slab[(v2 + hi * 8) * LDS_STRIDE + c] = fmaxf(f, 0.0f);
            }
        }
    }
    __syncthreads();
    if (active) {
        // ---- GIN layer 2: K = 128, back into acc ----
#pragma unroll
        for (int kt = 0; kt < 4; ++kt) {
            const int ko = kt * 32 + hi * 8;
            float u[8], v[8];
            load8(slab + lm * LDS_STRIDE + ko, u);
            load8(slab + lm * LDS_STRIDE + ko + 16, v);
            const bf16x16 a = pack16(u, v);
#pragma unroll
            for (int nt = 0; nt < 4; ++nt)
                acc[nt] = __builtin_amdgcn_wmma_f32_16x16x32_bf16(
                    false, a, false, W2v[(kt * 4 + nt) * 32 + lane],
                    (short)0, acc[nt], false, false);
        }
        // ---- combined bias + store ----
#pragma unroll
        for (int nt = 0; nt < 4; ++nt) {
            const int c = nt * 16 + lm;
            const float bb = btot[c];
#pragma unroll
            for (int v2 = 0; v2 < 8; ++v2)
                out[(size_t)(m0 + v2 + hi * 8) * 64 + c] = acc[nt][v2] + bb;
        }
    }
}

// ---------------------------------------------------------------------------
extern "C" void kernel_launch(void* const* d_in, const int* in_sizes, int n_in,
                              void* d_out, int out_size, void* d_ws, size_t ws_size,
                              hipStream_t stream)
{
    const float*     x    = (const float*)d_in[0];
    const float*     wts  = (const float*)d_in[1];
    const long long* ei   = (const long long*)d_in[2];
    const float*     ew   = (const float*)d_in[3];
    // d_in[4] = with_linear (unused; reference takes the false branch)
    const float* Wgcn  = (const float*)d_in[5];
    const float* bgcn  = (const float*)d_in[6];
    const float* Wss   = (const float*)d_in[7];
    const float* Wsn   = (const float*)d_in[8];
    const float* bsage = (const float*)d_in[9];
    const float* Wg1   = (const float*)d_in[10];
    const float* bg1   = (const float*)d_in[11];
    const float* Wg2   = (const float*)d_in[12];
    const float* bg2   = (const float*)d_in[13];
    const float* eps   = (const float*)d_in[14];
    const float* Wgc1  = (const float*)d_in[15];
    const float* Wgc2  = (const float*)d_in[16];
    const float* bgc   = (const float*)d_in[17];
    const float* Wmlp  = (const float*)d_in[18];
    const float* bmlp  = (const float*)d_in[19];
    float* out = (float*)d_out;

    const int N = in_sizes[0] / 64;
    const int E = in_sizes[3];

    // workspace layout (floats): wdeg[N] | gcn_pre[N*64] | agg[N*64] | btot[64] | b1f[128]
    // then 32B-aligned bf16 fragments: WBf[16384] | W1f[8192] | W2f[8192]
    float* ws      = (float*)d_ws;
    float* wdeg    = ws;
    float* gcn_pre = ws + (size_t)N;
    float* agg     = gcn_pre + (size_t)N * 64;
    float* btot    = agg + (size_t)N * 64;
    float* b1f     = btot + 64;
    size_t bfOff   = (((size_t)N * 129 + 192 + 7) / 8) * 8;   // 32B-aligned (in floats)
    __bf16* WBf = (__bf16*)(ws + bfOff);
    __bf16* W1f = WBf + 16384;
    __bf16* W2f = W1f + 8192;

    // 1) zero accumulators (wdeg, gcn_pre, agg are contiguous: N*129 floats)
    long n4 = ((long)N * 129 + 3) / 4;   // may spill into btot/b1f; prep rewrites them
    zero_kernel<<<1024, 256, 0, stream>>>((float4*)ws, n4);

    // 2) combine + swizzle weights into bf16 B-fragments
    prep_kernel<<<(32960 + 255) / 256, 256, 0, stream>>>(
        wts, Wgcn, bgcn, Wss, Wsn, bsage, Wg1, bg1, Wg2, bg2,
        Wgc1, Wgc2, bgc, Wmlp, bmlp, WBf, W1f, W2f, btot, b1f);

    // 3) weighted in-degree
    wdeg_kernel<<<(E + 255) / 256, 256, 0, stream>>>(ei, ew, wdeg, E);

    // 4) fused edge gather/scatter (single gather of x[src], two scatters)
    edge_kernel<<<(E + 15) / 16, 256, 0, stream>>>(x, ei, ew, wdeg, agg, gcn_pre, E);

    // 5) fused node-side WMMA kernel
    const int rowBlocks = (N + 15) / 16;
    const int gemmGrid  = (rowBlocks + GEMM_WAVES - 1) / GEMM_WAVES;
    gemm_kernel<<<gemmGrid, GEMM_WAVES * 32, 0, stream>>>(
        x, wdeg, gcn_pre, agg, WBf, W1f, W2f, btot, b1f, eps, out, N);
}
